// GridAttention_18339510353968
// MI455X (gfx1250) — matrix operand order repaired and checked by reading, not compiled, not run on hardware
//
#include <hip/hip_runtime.h>

typedef __bf16 bf16;
typedef __attribute__((ext_vector_type(16))) __bf16 bf16x16;
typedef __attribute__((ext_vector_type(8)))  __bf16 bf16x8;
typedef __attribute__((ext_vector_type(8)))  float  f32x8;

// Problem constants (from reference): b=8, l=2000, c=512, w=5, h=8, hd=64
#define NB    8
#define NL    2000
#define NC    512
#define NW    5
#define NH    8
#define HD    64
#define NG    40      // NB*NW groups
#define NT    400     // tokens per group (NL/NW)
#define NGH   320     // NG*NH
#define NTP   416     // NT padded to multiple of 32
#define KLSTR 72      // LDS stride for K rows ([j][d], d padded 64->72)
#define VLSTR 424     // LDS stride for V rows ([d][j], j padded 416->424)
#define ATT_SCALE 0.125f   // hd^-0.5

// ---------------- fragment helpers ----------------
__device__ __forceinline__ bf16x16 ldpair(const bf16* p0, const bf16* p1) {
  bf16x8 a = *(const bf16x8*)p0;
  bf16x8 b = *(const bf16x8*)p1;
  bf16x16 r;
#pragma unroll
  for (int i = 0; i < 8; ++i) { r[i] = a[i]; r[i + 8] = b[i]; }
  return r;
}
__device__ __forceinline__ bf16x16 ld16(const bf16* p) { return ldpair(p, p + 8); }

__device__ __forceinline__ f32x8 wmma_bf16(bf16x16 a, bf16x16 b, f32x8 c) {
  // 8 args: (neg_a, A, neg_b, B, c_mod, C, reuse_a, reuse_b)
  return __builtin_amdgcn_wmma_f32_16x16x32_bf16(false, a, false, b, (short)0, c,
                                                 false, false);
}

// wave-local LDS RAW sync (per-wave scratch; no block barrier needed)
__device__ __forceinline__ void wave_lds_sync() {
  __builtin_amdgcn_wave_barrier();
  asm volatile("s_wait_dscnt 0" ::: "memory");
  __builtin_amdgcn_wave_barrier();
}

// ---------------- prep kernels ----------------
// W (512 x 1536, f32, row-major K x N) -> Wt bf16 (1536 x 512, N-major)
__global__ void wcvt_kernel(const float* __restrict__ W, bf16* __restrict__ Wt) {
  int idx = blockIdx.x * 256 + threadIdx.x;
  if (idx >= NC * 3 * NC) return;
  int n = idx / NC, k = idx % NC;
  Wt[idx] = (bf16)W[k * (3 * NC) + n];
}

// x (8 x 2000 x 512 f32) -> xb bf16 [g=40][j=400][c=512] with grid unfold
__global__ void xcvt_kernel(const float* __restrict__ x, bf16* __restrict__ xb) {
  int idx = blockIdx.x * 256 + threadIdx.x;
  if (idx >= NB * NL * NC) return;
  int c = idx % NC;
  int t = idx / NC;
  int li = t % NL;
  int b0 = t / NL;
  int j = li / NW, wi = li % NW;
  xb[((size_t)(b0 * NW + wi) * NT + j) * NC + c] = (bf16)x[idx];
}

// ---------------- QKV GEMM: qkv = xb @ Wt^T + bias ----------------
// M=16000, N=1536, K=512.  One wave -> 32x64 tile (2 A-frags x 4 B-frags,
// 8 wmma per K-step); each 16x64 subtile is exactly one head of q/k/v.
#define GEMM_NCB 24   // 1536/64
#define GEMM_NRB 500  // 16000/32
__global__ void __launch_bounds__(128)
qkv_gemm_kernel(const bf16* __restrict__ xb, const bf16* __restrict__ Wt,
                const float* __restrict__ bias,
                bf16* __restrict__ qws, bf16* __restrict__ kws,
                bf16* __restrict__ vtws) {
  const int lane = threadIdx.x & 31;
  const int wid = blockIdx.x * 4 + (threadIdx.x >> 5);
  if (wid >= GEMM_NRB * GEMM_NCB) return;
  const int rb32 = wid / GEMM_NCB;
  const int cb = wid % GEMM_NCB;
  const int hi = lane >> 4, ln = lane & 15;
  const int base = 8 * hi;

  const bf16* arow0 = xb + (size_t)(rb32 * 32 + ln) * NC;        // rows 0..15
  const bf16* arow1 = arow0 + (size_t)16 * NC;                   // rows 16..31

  f32x8 acc[2][4];
#pragma unroll
  for (int mr = 0; mr < 2; ++mr)
#pragma unroll
    for (int nd = 0; nd < 4; ++nd)
#pragma unroll
      for (int r = 0; r < 8; ++r) acc[mr][nd][r] = 0.0f;

  for (int kk = 0; kk < NC; kk += 32) {
    if (kk + 64 < NC) {     // gfx1250 global_prefetch_b8, 2 K-steps ahead
      __builtin_prefetch(arow0 + kk + 64, 0, 1);
      __builtin_prefetch(arow1 + kk + 64, 0, 1);
    }
    bf16x16 a0 = ldpair(arow0 + kk + 16 * hi, arow0 + kk + 16 * hi + 8);
    bf16x16 a1 = ldpair(arow1 + kk + 16 * hi, arow1 + kk + 16 * hi + 8);
#pragma unroll
    for (int nd = 0; nd < 4; ++nd) {
      int n = cb * 64 + nd * 16 + ln;
      bf16x16 b = ld16(Wt + (size_t)n * NC + kk + 16 * hi);
      acc[0][nd] = wmma_bf16(a0, b, acc[0][nd]);
      acc[1][nd] = wmma_bf16(a1, b, acc[1][nd]);
    }
  }

#pragma unroll
  for (int mr = 0; mr < 2; ++mr) {
    const int row0 = rb32 * 32 + mr * 16;   // 16-row subtile start (400%16==0)
    const int g = row0 / NT;
    const int j0 = row0 % NT;
#pragma unroll
    for (int nd = 0; nd < 4; ++nd) {
      const int col = cb * 64 + nd * 16 + ln;   // output channel in [0,1536)
      const float bv = bias[col];
      const int sect = col >> 9;                // 0=q 1=k 2=v
      const int ch = col & 511;
      const int h = ch >> 6, d = ch & 63;
      if (sect == 2) {
        // v stored transposed: vt[g][h][d][j], 8 consecutive j -> one 16B store
        bf16x8 pk;
#pragma unroll
        for (int r = 0; r < 8; ++r) pk[r] = (bf16)(acc[mr][nd][r] + bv);
        *(bf16x8*)(vtws + ((size_t)(g * NH + h) * HD + d) * NT + j0 + 8 * hi) = pk;
      } else {
        bf16* dst = (sect == 0) ? qws : kws;    // [g][h][j][d]
#pragma unroll
        for (int r = 0; r < 8; ++r) {
          int j = j0 + 8 * hi + r;              // C-layout: element r -> row r+8*hi
          dst[((size_t)(g * NH + h) * NT + j) * HD + d] = (bf16)(acc[mr][nd][r] + bv);
        }
      }
    }
  }
}

// ---------------- fused attention: softmax(QK^T*s) @ V, fold back ----------------
__global__ void __launch_bounds__(128)
attn_kernel(const bf16* __restrict__ qws, const bf16* __restrict__ kws,
            const bf16* __restrict__ vtws, float* __restrict__ out) {
  __shared__ bf16 kl[NTP * KLSTR];          // K   [j][d]  (59.9 KB)
  __shared__ bf16 vl[HD * VLSTR];           // V^T [d][j]  (54.3 KB)
  __shared__ float pl[4][16][33];           // per-wave P relayout scratch (8.4 KB)

  const int gh = blockIdx.x;                // 0..319
  const int g = gh >> 3, h = gh & 7;
  const int b0 = g / NW, wi = g % NW;
  const int tid = threadIdx.x;
  const int lane = tid & 31, wv = tid >> 5;
  const int hi = lane >> 4, ln = lane & 15;
  const int base = 8 * hi;

  const bf16* kg = kws + (size_t)gh * NT * HD;
  const bf16* vg = vtws + (size_t)gh * HD * NT;

  bf16x8 zed;
#pragma unroll
  for (int i = 0; i < 8; ++i) zed[i] = (bf16)0.0f;

  // stage K: 400 rows of 64, into padded stride-72 rows
  for (int i = tid; i < NT * 8; i += 128) {
    int j = i >> 3, c = i & 7;
    ((bf16x8*)kl)[j * (KLSTR / 8) + c] = ((const bf16x8*)kg)[i];
  }
  for (int i = tid; i < 16 * (KLSTR / 8); i += 128)      // zero rows 400..415
    ((bf16x8*)(kl + NT * KLSTR))[i] = zed;
  // stage V^T: 64 rows of 400, into padded stride-424 rows; zero j in [400,424)
  for (int i = tid; i < HD * 50; i += 128) {
    int d = i / 50, c = i % 50;
    ((bf16x8*)(vl + d * VLSTR))[c] = ((const bf16x8*)(vg + d * NT))[c];
  }
  for (int i = tid; i < HD * 3; i += 128) {
    int d = i / 3, c = i % 3;
    ((bf16x8*)(vl + d * VLSTR + NT))[c] = zed;
  }
  __syncthreads();

  // each wave owns 16-row Q blocks: 25 blocks over 4 waves
  for (int blk = wv; blk < 25; blk += 4) {
    const bf16* qrow = qws + ((size_t)gh * NT + blk * 16 + ln) * HD;
    if (blk + 4 < 25)   // prefetch next Q row-block (global_prefetch_b8)
      __builtin_prefetch(qrow + (size_t)64 * HD, 0, 1);
    bf16x16 qa0 = ldpair(qrow + 16 * hi, qrow + 16 * hi + 8);        // K(d)=0..31
    bf16x16 qa1 = ldpair(qrow + 32 + 16 * hi, qrow + 40 + 16 * hi);   // K(d)=32..63

    float mrow[8], lrow[8];
    f32x8 o[4];
#pragma unroll
    for (int r = 0; r < 8; ++r) { mrow[r] = -1e30f; lrow[r] = 0.0f; }
#pragma unroll
    for (int nd = 0; nd < 4; ++nd)
#pragma unroll
      for (int r = 0; r < 8; ++r) o[nd][r] = 0.0f;

    for (int jc = 0; jc < 13; ++jc) {
      const int j0 = jc * 32;
      f32x8 s0, s1;
#pragma unroll
      for (int r = 0; r < 8; ++r) { s0[r] = 0.0f; s1[r] = 0.0f; }

      // scores: two 16x16 tiles, each K=64 over two wmma
      const bf16* kr0 = kl + (size_t)(j0 + ln) * KLSTR;
      s0 = wmma_bf16(qa0, ld16(kr0 + 16 * hi), s0);
      s0 = wmma_bf16(qa1, ld16(kr0 + 32 + 16 * hi), s0);
      const bf16* kr1 = kl + (size_t)(j0 + 16 + ln) * KLSTR;
      s1 = wmma_bf16(qa0, ld16(kr1 + 16 * hi), s1);
      s1 = wmma_bf16(qa1, ld16(kr1 + 32 + 16 * hi), s1);

      const bool mask_hi = (jc == 12);       // cols 400..415 invalid
#pragma unroll
      for (int r = 0; r < 8; ++r) {
        s0[r] *= ATT_SCALE;
        s1[r] = mask_hi ? -1e30f : s1[r] * ATT_SCALE;
      }

      // online softmax (row = r + 8*hi; rows live across the 16-lane half)
      float cmax[8];
#pragma unroll
      for (int r = 0; r < 8; ++r) cmax[r] = fmaxf(s0[r], s1[r]);
#pragma unroll
      for (int m = 1; m < 16; m <<= 1)
#pragma unroll
        for (int r = 0; r < 8; ++r)
          cmax[r] = fmaxf(cmax[r], __shfl_xor(cmax[r], m, 32));

      float alpha[8], psum[8];
#pragma unroll
      for (int r = 0; r < 8; ++r) {
        float mn = fmaxf(mrow[r], cmax[r]);
        alpha[r] = __expf(mrow[r] - mn);
        s0[r] = __expf(s0[r] - mn);
        s1[r] = __expf(s1[r] - mn);
        psum[r] = s0[r] + s1[r];
        mrow[r] = mn;
      }
#pragma unroll
      for (int m = 1; m < 16; m <<= 1)
#pragma unroll
        for (int r = 0; r < 8; ++r) psum[r] += __shfl_xor(psum[r], m, 32);
#pragma unroll
      for (int r = 0; r < 8; ++r) lrow[r] = lrow[r] * alpha[r] + psum[r];
#pragma unroll
      for (int nd = 0; nd < 4; ++nd)
#pragma unroll
        for (int r = 0; r < 8; ++r) o[nd][r] *= alpha[r];

      // relayout P: C-layout -> A-fragment layout via per-wave LDS scratch
#pragma unroll
      for (int r = 0; r < 8; ++r) {
        pl[wv][r + 8 * hi][ln] = s0[r];
        pl[wv][r + 8 * hi][16 + ln] = s1[r];
      }
      wave_lds_sync();
      bf16x16 pa;
#pragma unroll
      for (int i = 0; i < 8; ++i) {
        pa[i] = (bf16)pl[wv][ln][16 * hi + i];
        pa[8 + i] = (bf16)pl[wv][ln][16 * hi + 8 + i];
      }
      wave_lds_sync();

      // o += P @ V (4 d-tiles, K=32 j's)
#pragma unroll
      for (int nd = 0; nd < 4; ++nd) {
        const bf16* vr = vl + (size_t)(nd * 16 + ln) * VLSTR + j0 + 16 * hi;
        o[nd] = wmma_bf16(pa, ld16(vr), o[nd]);
      }
    }

    // normalize and fold back: token l = (blk*16 + M)*5 + wi, ch = h*64 + d
    float linv[8];
#pragma unroll
    for (int r = 0; r < 8; ++r) linv[r] = 1.0f / lrow[r];
#pragma unroll
    for (int nd = 0; nd < 4; ++nd) {
#pragma unroll
      for (int r = 0; r < 8; ++r) {
        int jtok = blk * 16 + r + 8 * hi;
        int lidx = jtok * NW + wi;
        int ch = h * HD + nd * 16 + ln;
        out[((size_t)b0 * NL + lidx) * NC + ch] = o[nd][r] * linv[r];
      }
    }
  }
}

// ---------------- host launcher ----------------
extern "C" void kernel_launch(void* const* d_in, const int* in_sizes, int n_in,
                              void* d_out, int out_size, void* d_ws, size_t ws_size,
                              hipStream_t stream) {
  const float* x = (const float*)d_in[0];      // (8,2000,512)
  const float* W = (const float*)d_in[1];      // (512,1536)
  const float* bias = (const float*)d_in[2];   // (1536,)
  float* out = (float*)d_out;                  // (8,2000,512)

  char* ws = (char*)d_ws;
  const size_t xb_off = 0;
  const size_t xb_sz = (size_t)NG * NT * NC * 2;          // 16,384,000
  const size_t wt_off = xb_off + xb_sz;
  const size_t wt_sz = (size_t)3 * NC * NC * 2;           //  1,572,864
  const size_t q_off = wt_off + wt_sz;
  const size_t qkv_each = (size_t)NGH * NT * HD * 2;      // 16,384,000
  const size_t k_off = q_off + qkv_each;
  const size_t v_off = k_off + qkv_each;

  bf16* xb = (bf16*)(ws + xb_off);
  bf16* Wt = (bf16*)(ws + wt_off);
  bf16* qws = (bf16*)(ws + q_off);
  bf16* kws = (bf16*)(ws + k_off);
  bf16* vtws = (bf16*)(ws + v_off);

  wcvt_kernel<<<(NC * 3 * NC + 255) / 256, 256, 0, stream>>>(W, Wt);
  xcvt_kernel<<<(NB * NL * NC + 255) / 256, 256, 0, stream>>>(x, xb);
  qkv_gemm_kernel<<<(GEMM_NRB * GEMM_NCB) / 4, 128, 0, stream>>>(xb, Wt, bias,
                                                                 qws, kws, vtws);
  attn_kernel<<<NGH, 128, 0, stream>>>(qws, kws, vtws, out);
}
